// NeuroSymbolicWaveNet_54228257079953
// MI455X (gfx1250) — compile-verified, matrix operand order checked by reading, not run on hardware
//
#include <hip/hip_runtime.h>
#include <cstddef>

// ---------------------------------------------------------------------------
// NeuroSymbolicWaveNet on gfx1250 (MI455X), wave32 + WMMA bf16.
//   B=2 S=1024 DIM=2048 H=16 HD=128 MODES=16 DEG=5 DEPTH=4 MD=32 base=32
// All big matmuls -> v_wmma_f32_16x16x32_bf16 (f32 accum).
// GEMM: 128x128 macro-tile, 8 waves (2x4), 8 WMMA / 6 frag-loads per K-step,
//       register double-buffered global->LDS staging.
// ---------------------------------------------------------------------------

typedef __attribute__((ext_vector_type(16))) __bf16        v16bf;
typedef __attribute__((ext_vector_type(8)))  float         v8f;
typedef __attribute__((ext_vector_type(4)))  unsigned int  u32x4;
typedef __attribute__((ext_vector_type(4)))  float         f32x4;

struct FragBF { u32x4 lo, hi; };   // 32 bytes == v16bf

__device__ __forceinline__ unsigned short f2bf(float f) {
  union { float f; unsigned int u; } v; v.f = f;
  unsigned int u = v.u;
  unsigned int r = u + 0x7fffu + ((u >> 16) & 1u);   // round-to-nearest-even
  return (unsigned short)(r >> 16);
}

__device__ __forceinline__ v8f wmma_bf16(v16bf a, v16bf b, v8f c) {
  return __builtin_amdgcn_wmma_f32_16x16x32_bf16(false, a, false, b, (short)0, c,
                                                 false, false);
}

// ---------------------------------------------------------------------------
// 1) Symbolic path: x[b,s,:] = vec( G_t[g0] @ G_t[g1] @ G_t[g2] ), t=0,1
// ---------------------------------------------------------------------------
__global__ __launch_bounds__(256)
void sym_kernel(const int* __restrict__ ids, const int* __restrict__ tmap,
                const float* __restrict__ gens, float* __restrict__ x) {
  const int bs = blockIdx.x;          // 0..2047 == b*1024+s
  const int tid = threadIdx.x;
  __shared__ float Al[32][33], Bl[32][33], Pl[32][33];
  const int id = ids[bs];
  const int g0 = tmap[id * 3 + 0], g1 = tmap[id * 3 + 1], g2 = tmap[id * 3 + 2];
  for (int t = 0; t < 2; ++t) {
    const float* A  = gens + (size_t)(t * 32 + g0) * 1024;
    const float* Bm = gens + (size_t)(t * 32 + g1) * 1024;
    const float* Cm = gens + (size_t)(t * 32 + g2) * 1024;
    __syncthreads();
    for (int j = 0; j < 4; ++j) {
      int li = tid + 256 * j, i = li >> 5, c = li & 31;
      Al[i][c] = A[li]; Bl[i][c] = Bm[li];
    }
    __syncthreads();
    for (int j = 0; j < 4; ++j) {
      int li = tid + 256 * j, i = li >> 5, c = li & 31;
      float s = 0.f;
      #pragma unroll
      for (int k = 0; k < 32; ++k) s += Al[i][k] * Bl[k][c];
      Pl[i][c] = s;
    }
    __syncthreads();
    for (int j = 0; j < 4; ++j) {
      int li = tid + 256 * j, i = li >> 5, c = li & 31;
      Bl[i][c] = Cm[li];
    }
    __syncthreads();
    for (int j = 0; j < 4; ++j) {
      int li = tid + 256 * j, i = li >> 5, c = li & 31;
      float s = 0.f;
      #pragma unroll
      for (int k = 0; k < 32; ++k) s += Pl[i][k] * Bl[k][c];
      x[(size_t)bs * 2048 + t * 1024 + li] = s;
    }
  }
}

// ---------------------------------------------------------------------------
// 2) kan_coeffs (i,o,d) f32 -> Bkan[(d*2048+i)*2048+o] bf16  (GEMM B matrix)
// ---------------------------------------------------------------------------
__global__ __launch_bounds__(256)
void kan_prep_kernel(const float* __restrict__ C, unsigned short* __restrict__ Bk) {
  size_t idx = (size_t)blockIdx.x * 256 + threadIdx.x;  // i*2048+o
  int o = (int)(idx & 2047);
  int i = (int)(idx >> 11);
  const float* p = C + idx * 5;
  #pragma unroll
  for (int d = 0; d < 5; ++d)
    Bk[(size_t)d * 2048 * 2048 + (size_t)i * 2048 + o] = f2bf(p[d]);
}

// ---------------------------------------------------------------------------
// 3) DFT: xf[b,m,i] = sum_s x[b,s,i] * e^{-2πi m s/1024}, m<16
// ---------------------------------------------------------------------------
__global__ __launch_bounds__(256)
void dft_kernel(const float* __restrict__ x, float* __restrict__ xfr,
                float* __restrict__ xfi) {
  const int bid = blockIdx.x;           // b*128 + m*8 + ic
  const int ic = bid & 7, m = (bid >> 3) & 15, b = bid >> 7;
  const int i = ic * 256 + threadIdx.x;
  __shared__ float tc[256], ts[256];
  float ar = 0.f, ai = 0.f;
  for (int s0 = 0; s0 < 1024; s0 += 256) {
    __syncthreads();
    {
      float th = -6.283185307179586f * (float)m * (float)(s0 + threadIdx.x) / 1024.0f;
      float sv, cv; sincosf(th, &sv, &cv);
      tc[threadIdx.x] = cv; ts[threadIdx.x] = sv;
    }
    __syncthreads();
    for (int ss = 0; ss < 256; ++ss) {
      float xv = x[((size_t)(b * 1024 + s0 + ss)) * 2048 + i];
      ar += xv * tc[ss];
      ai += xv * ts[ss];
    }
  }
  xfr[((size_t)(b * 16 + m)) * 2048 + i] = ar;
  xfi[((size_t)(b * 16 + m)) * 2048 + i] = ai;
}

// ---------------------------------------------------------------------------
// 4) Spectral mix: om[b,m,o] = sum_i xf[b,m,i] * (wr+j wi)[i,o,m]
//    One thread per o, all 16 modes at once -> wr[i,o,0..15] is a coalesced 64B read.
// ---------------------------------------------------------------------------
__global__ __launch_bounds__(64)
void spec_kernel(const float* __restrict__ xfr, const float* __restrict__ xfi,
                 const float* __restrict__ wr, const float* __restrict__ wi,
                 float* __restrict__ omr, float* __restrict__ omi) {
  const int tid = threadIdx.x;
  const int o = blockIdx.x * 64 + tid;
  __shared__ float sr[32][65], si[32][65];   // rows bm = b*16+m
  float aR[2][16] = {}, aI[2][16] = {};
  for (int i0 = 0; i0 < 2048; i0 += 64) {
    __syncthreads();
    for (int j = 0; j < 32; ++j) {
      int li = tid + 64 * j;                 // 0..2047
      int bm = li >> 6, ii = li & 63;
      sr[bm][ii] = xfr[(size_t)bm * 2048 + i0 + ii];
      si[bm][ii] = xfi[(size_t)bm * 2048 + i0 + ii];
    }
    __syncthreads();
    for (int ii = 0; ii < 64; ++ii) {
      size_t base = ((size_t)(i0 + ii) * 2048 + o) * 16;
      const f32x4* wr4 = (const f32x4*)(wr + base);
      const f32x4* wi4 = (const f32x4*)(wi + base);
      float wrv[16], wiv[16];
      #pragma unroll
      for (int q = 0; q < 4; ++q) {
        f32x4 a = wr4[q], c = wi4[q];
        wrv[4 * q] = a.x; wrv[4 * q + 1] = a.y; wrv[4 * q + 2] = a.z; wrv[4 * q + 3] = a.w;
        wiv[4 * q] = c.x; wiv[4 * q + 1] = c.y; wiv[4 * q + 2] = c.z; wiv[4 * q + 3] = c.w;
      }
      #pragma unroll
      for (int bb = 0; bb < 2; ++bb) {
        #pragma unroll
        for (int mm = 0; mm < 16; ++mm) {
          float xr = sr[bb * 16 + mm][ii], xi = si[bb * 16 + mm][ii];
          aR[bb][mm] += xr * wrv[mm] - xi * wiv[mm];
          aI[bb][mm] += xr * wiv[mm] + xi * wrv[mm];
        }
      }
    }
  }
  #pragma unroll
  for (int bb = 0; bb < 2; ++bb)
    #pragma unroll
    for (int mm = 0; mm < 16; ++mm) {
      omr[((size_t)(bb * 16 + mm)) * 2048 + o] = aR[bb][mm];
      omi[((size_t)(bb * 16 + mm)) * 2048 + o] = aI[bb][mm];
    }
}

// ---------------------------------------------------------------------------
// 5) Generic bf16 WMMA GEMM:  C[M,N] (+)= A[M,K] * B[K,N]
//    128x128 macro-tile, 8 waves (2 along M x 4 along N), each wave 64x32:
//    8 WMMA per K-step per wave. Register double-buffered staging.
//    CHEB:    A element = T_d(clamp(tanh(x[m,i]))),  kk = d*lda + i
//    BLAYOUT: 1 -> Bsrc = f32 weights laid out [N][K] (projection W[e,d])
//             0 -> Bsrc = bf16 [K][N] (prepped kan tensor)
//    ACC:     1 -> accumulate into C (residual add)
// ---------------------------------------------------------------------------
template <int CHEB, int BLAYOUT, int ACC>
__global__ __launch_bounds__(256)
void gemm_bf16_kernel(const float* __restrict__ A, const void* __restrict__ Bsrc,
                      float* __restrict__ C, int M, int N, int K, int lda) {
  __shared__ alignas(16) unsigned short Al[128][40];
  __shared__ alignas(16) unsigned short Bt[128][40];   // [n][k]
  const int tid = threadIdx.x;
  const int wid = tid >> 5, lane = tid & 31;
  const int hi = lane >> 4, ln = lane & 15;
  const int mw = wid & 1, nw = wid >> 1;               // 2 x 4 wave grid
  const int m0 = blockIdx.y * 128, n0 = blockIdx.x * 128;
  v8f acc[4][2] = {};

  float aReg[16];
  float bRegF[16];
  unsigned short bRegH[16];

  auto loadA = [&](int k0) {
    #pragma unroll
    for (int j = 0; j < 16; ++j) {
      int li = tid + 256 * j;
      int r = li >> 5, c = li & 31;
      int kk = k0 + c;
      int i = kk;
      if (CHEB) { int d = kk / lda; i = kk - d * lda; }
      aReg[j] = A[(size_t)(m0 + r) * lda + i];
    }
  };
  auto loadB = [&](int k0) {
    if (BLAYOUT == 1) {
      const float* W = (const float*)Bsrc;
      #pragma unroll
      for (int j = 0; j < 16; ++j) {
        int li = tid + 256 * j;
        int n = li >> 5, c = li & 31;
        bRegF[j] = W[(size_t)(n0 + n) * K + k0 + c];
      }
    } else {
      const unsigned short* Bk = (const unsigned short*)Bsrc;
      #pragma unroll
      for (int j = 0; j < 16; ++j) {
        int li = tid + 256 * j;
        int kk = li >> 7, n = li & 127;
        bRegH[j] = Bk[(size_t)(k0 + kk) * N + n0 + n];
      }
    }
  };
  auto storeTiles = [&](int k0) {
    #pragma unroll
    for (int j = 0; j < 16; ++j) {                    // A: f32 -> (Chebyshev) -> bf16
      int li = tid + 256 * j;
      int r = li >> 5, c = li & 31;
      float val = aReg[j];
      if (CHEB) {
        int kk = k0 + c;
        int d = kk / lda;
        float t = tanhf(val);
        t = fminf(fmaxf(t, -1.0f + 1e-6f), 1.0f - 1e-6f);
        float Td;
        if (d == 0) Td = 1.0f;
        else {
          float T0 = 1.0f, T1 = t;
          for (int q = 2; q <= d; ++q) { float Tn = 2.0f * t * T1 - T0; T0 = T1; T1 = Tn; }
          Td = T1;
        }
        val = Td;
      }
      Al[r][c] = f2bf(val);
    }
    if (BLAYOUT == 1) {
      #pragma unroll
      for (int j = 0; j < 16; ++j) {
        int li = tid + 256 * j;
        int n = li >> 5, c = li & 31;
        Bt[n][c] = f2bf(bRegF[j]);
      }
    } else {
      #pragma unroll
      for (int j = 0; j < 16; ++j) {
        int li = tid + 256 * j;
        int kk = li >> 7, n = li & 127;
        Bt[n][kk] = bRegH[j];
      }
    }
  };

  loadA(0); loadB(0);
  for (int k0 = 0; k0 < K; k0 += 32) {
    storeTiles(k0);
    __syncthreads();
    if (k0 + 32 < K) { loadA(k0 + 32); loadB(k0 + 32); }   // overlap with WMMA below
    FragBF bfr[2];
    #pragma unroll
    for (int t = 0; t < 2; ++t) {
      const u32x4* brow = (const u32x4*)(&Bt[nw * 32 + t * 16 + ln][0]);
      bfr[t].lo = brow[hi]; bfr[t].hi = brow[2 + hi];
    }
    #pragma unroll
    for (int mt = 0; mt < 4; ++mt) {
      const u32x4* arow = (const u32x4*)(&Al[mw * 64 + mt * 16 + ln][0]);
      FragBF af; af.lo = arow[hi]; af.hi = arow[2 + hi];
      v16bf av = __builtin_bit_cast(v16bf, af);
      #pragma unroll
      for (int t = 0; t < 2; ++t)
        acc[mt][t] = wmma_bf16(av, __builtin_bit_cast(v16bf, bfr[t]), acc[mt][t]);
    }
    __syncthreads();
  }
  #pragma unroll
  for (int mt = 0; mt < 4; ++mt) {
    const int rbase = m0 + mw * 64 + mt * 16 + hi * 8;
    #pragma unroll
    for (int t = 0; t < 2; ++t) {
      int col = n0 + nw * 32 + t * 16 + ln;
      #pragma unroll
      for (int r = 0; r < 8; ++r) {
        size_t idx = (size_t)(rbase + r) * N + col;
        float v = acc[mt][t][r];
        if (ACC) v += C[idx];
        C[idx] = v;
      }
    }
  }
}

// ---------------------------------------------------------------------------
// 6) x = gelu( irfft16(om)[b,s,o] + x2[b,s,o] )
// ---------------------------------------------------------------------------
__global__ __launch_bounds__(256)
void ifft_gelu_kernel(const float* __restrict__ omr, const float* __restrict__ omi,
                      const float* __restrict__ x2, float* __restrict__ x) {
  size_t idx = (size_t)blockIdx.x * 256 + threadIdx.x;
  int o = (int)(idx & 2047);
  int s = (int)((idx >> 11) & 1023);
  int b = (int)(idx >> 21);
  float ph = 6.283185307179586f * (float)s / 1024.0f;
  float c1, s1; sincosf(ph, &s1, &c1);
  float acc = omr[((size_t)(b * 16)) * 2048 + o];   // DC: real part only (irfft)
  float cm = c1, sm = s1;
  for (int m = 1; m < 16; ++m) {
    float yr = omr[((size_t)(b * 16 + m)) * 2048 + o];
    float yi = omi[((size_t)(b * 16 + m)) * 2048 + o];
    acc += 2.0f * (yr * cm - yi * sm);
    float cn = cm * c1 - sm * s1;
    float sn = sm * c1 + cm * s1;
    cm = cn; sm = sn;
  }
  float x1 = acc * (1.0f / 1024.0f);
  float v = x1 + x2[idx];
  x[idx] = 0.5f * v * (1.0f + erff(v * 0.7071067811865475f));
}

// ---------------------------------------------------------------------------
// 7) RMSNorm (per row of 2048)
// ---------------------------------------------------------------------------
__global__ __launch_bounds__(256)
void rmsnorm_kernel(const float* __restrict__ x, const float* __restrict__ w,
                    float* __restrict__ out) {
  const int row = blockIdx.x;
  const float* xr = x + (size_t)row * 2048;
  __shared__ float red[256];
  float s = 0.f;
  for (int j = 0; j < 8; ++j) { float v = xr[threadIdx.x + 256 * j]; s += v * v; }
  red[threadIdx.x] = s; __syncthreads();
  for (int st = 128; st > 0; st >>= 1) {
    if (threadIdx.x < st) red[threadIdx.x] += red[threadIdx.x + st];
    __syncthreads();
  }
  float inv = rsqrtf(red[0] * (1.0f / 2048.0f) + 1e-6f);
  for (int j = 0; j < 8; ++j) {
    int c = threadIdx.x + 256 * j;
    out[(size_t)row * 2048 + c] = xr[c] * inv * w[c];
  }
}

// ---------------------------------------------------------------------------
// 8) RoPE in place on (b,s,h,128)
// ---------------------------------------------------------------------------
__global__ __launch_bounds__(256)
void rope_kernel(float* __restrict__ q) {
  size_t idx = (size_t)blockIdx.x * 256 + threadIdx.x;  // b*1024*16*64
  int j = (int)(idx & 63);
  int h = (int)((idx >> 6) & 15);
  int s = (int)((idx >> 10) & 1023);
  int b = (int)(idx >> 20);
  float invf = expf(-((float)(2 * j) / 128.0f) * logf(10000.0f));
  float th = (float)s * invf;
  float c, sn; sincosf(th, &sn, &c);
  size_t base = ((size_t)(b * 1024 + s)) * 2048 + h * 128;
  float a = q[base + j], bq = q[base + j + 64];
  q[base + j]      = a * c  - bq * sn;
  q[base + j + 64] = bq * c + a  * sn;
}

// ---------------------------------------------------------------------------
// 9) Attention: tanh-kernel weights, linear normalization -> single pass.
//    block = (b,h,8 query-tiles); wave = one 16-query tile; keys in chunks of 32.
// ---------------------------------------------------------------------------
__global__ __launch_bounds__(256)
void attn_kernel(const float* __restrict__ qbuf, const float* __restrict__ kbuf,
                 const float* __restrict__ vbuf, const float* __restrict__ comp,
                 const float* __restrict__ phse, float* __restrict__ ao) {
  const int bid = blockIdx.x;
  const int qg = bid & 7, h = (bid >> 3) & 15, b = bid >> 7;
  const int tid = threadIdx.x;
  const int wid = tid >> 5, lane = tid & 31;
  const int hi = lane >> 4, ln = lane & 15;
  const int qt = qg * 8 + wid;
  const int sq = qt * 16;

  __shared__ alignas(16) unsigned short Klds[32][136];     // [key][hd]
  __shared__ alignas(16) unsigned short Vlds[128][40];     // [hd][key] (transposed)
  __shared__ alignas(16) unsigned short Qlds[8][16][136];  // per-wave
  __shared__ alignas(16) unsigned short Wlds[8][16][40];   // per-wave weight tile

  const float scale = 0.08838834764831845f;                // 1/sqrt(128)
  const float tau = log1pf(expf(comp[h])) + 1e-6f;
  const float ph = phse[h];

  for (int j = 0; j < 64; ++j) {                           // stage Q (wave-local)
    int li = lane + 32 * j;
    int r = li >> 7, c = li & 127;
    Qlds[wid][r][c] = f2bf(qbuf[((size_t)(b * 1024 + sq + r)) * 2048 + h * 128 + c]);
  }
  __syncthreads();
  FragBF qf[4];
  {
    const u32x4* rowp = (const u32x4*)(&Qlds[wid][ln][0]);
    #pragma unroll
    for (int kk = 0; kk < 4; ++kk) { qf[kk].lo = rowp[kk * 4 + hi]; qf[kk].hi = rowp[kk * 4 + 2 + hi]; }
  }

  v8f oacc[8] = {};
  float wsum[8] = {};
  const int nchunk = (qg + 1) * 4;                         // covers block's causal range
  for (int ch = 0; ch < nchunk; ++ch) {
    const int j0 = ch * 32;
    for (int j = 0; j < 16; ++j) {                         // stage K & V (32x128)
      int li = tid + 256 * j;
      int r = li >> 7, c = li & 127;
      size_t g = ((size_t)(b * 1024 + j0 + r)) * 2048 + h * 128 + c;
      Klds[r][c] = f2bf(kbuf[g]);
      Vlds[c][r] = f2bf(vbuf[g]);
    }
    __syncthreads();
    const bool active = (j0 <= sq + 15);
    if (active) {
      v8f sacc[2] = {};
      #pragma unroll
      for (int kk = 0; kk < 4; ++kk) {                     // scores Q * K^T
        v16bf av = __builtin_bit_cast(v16bf, qf[kk]);
        #pragma unroll
        for (int t = 0; t < 2; ++t) {
          const u32x4* rowp = (const u32x4*)(&Klds[t * 16 + ln][0]);
          FragBF bf_; bf_.lo = rowp[kk * 4 + hi]; bf_.hi = rowp[kk * 4 + 2 + hi];
          sacc[t] = wmma_bf16(av, __builtin_bit_cast(v16bf, bf_), sacc[t]);
        }
      }
      #pragma unroll
      for (int t = 0; t < 2; ++t) {                        // w = 1 - tanh^2(e), causal
        #pragma unroll
        for (int r = 0; r < 8; ++r) {
          int irow = sq + hi * 8 + r;
          int jcol = j0 + t * 16 + ln;
          float w = 0.0f;
          if (jcol <= irow) {
            float sc = sacc[t][r] * scale;
            float e = (sc - ph) / tau;
            float th = tanhf(e);
            w = 1.0f - th * th;
          }
          wsum[r] += w;
          Wlds[wid][hi * 8 + r][t * 16 + ln] = f2bf(w);
        }
      }
      asm volatile("s_wait_dscnt 0" ::: "memory");         // wave-local LDS RAW
      FragBF wf;
      { const u32x4* rowp = (const u32x4*)(&Wlds[wid][ln][0]);
        wf.lo = rowp[hi]; wf.hi = rowp[2 + hi]; }
      v16bf wv = __builtin_bit_cast(v16bf, wf);
      #pragma unroll
      for (int nt = 0; nt < 8; ++nt) {                     // out += W * V
        const u32x4* rowp = (const u32x4*)(&Vlds[nt * 16 + ln][0]);
        FragBF bf_; bf_.lo = rowp[hi]; bf_.hi = rowp[2 + hi];
        oacc[nt] = wmma_bf16(wv, __builtin_bit_cast(v16bf, bf_), oacc[nt]);
      }
    }
    __syncthreads();
  }
  #pragma unroll
  for (int r = 0; r < 8; ++r) {                            // row-sum across 16-lane half
    float v = wsum[r];
    for (int m = 1; m <= 8; m <<= 1) v += __shfl_xor(v, m, 32);
    wsum[r] = v;
  }
  #pragma unroll
  for (int nt = 0; nt < 8; ++nt)
    #pragma unroll
    for (int r = 0; r < 8; ++r) {
      int irow = sq + hi * 8 + r;
      float v = oacc[nt][r] / (wsum[r] + 1e-6f);
      ao[((size_t)(b * 1024 + irow)) * 2048 + h * 128 + nt * 16 + ln] = v;
    }
}

// ---------------------------------------------------------------------------
// Host orchestration
// ---------------------------------------------------------------------------
extern "C" void kernel_launch(void* const* d_in, const int* in_sizes, int n_in,
                              void* d_out, int out_size, void* d_ws, size_t ws_size,
                              hipStream_t stream) {
  (void)in_sizes; (void)n_in; (void)out_size; (void)ws_size;
  const int*   ids   = (const int*)d_in[0];
  const int*   tmap  = (const int*)d_in[1];
  const float* gens  = (const float*)d_in[2];
  const float* swr   = (const float*)d_in[3];
  const float* swi   = (const float*)d_in[4];
  const float* kanc  = (const float*)d_in[5];
  const float* normw = (const float*)d_in[6];
  const float* qw    = (const float*)d_in[7];
  const float* kw    = (const float*)d_in[8];
  const float* vw    = (const float*)d_in[9];
  const float* ow    = (const float*)d_in[10];
  const float* comp  = (const float*)d_in[11];
  const float* phs   = (const float*)d_in[12];
  const float* fnw   = (const float*)d_in[13];
  float* out = (float*)d_out;

  char* ws = (char*)d_ws;                      // ~154 MB total
  float* x   = (float*)(ws + 0);               // 16 MB  (B,S,DIM)
  float* x2  = (float*)(ws + 16777216);        // 16 MB
  float* hb  = (float*)(ws + 33554432);        // 16 MB
  float* qb  = (float*)(ws + 50331648);        // 16 MB
  float* kb  = (float*)(ws + 67108864);        // 16 MB
  float* vb  = (float*)(ws + 83886080);        // 16 MB
  float* ao  = (float*)(ws + 100663296);       // 16 MB
  float* xfr = (float*)(ws + 117440512);       // 256 KB
  float* xfi = (float*)(ws + 117702656);
  float* omr = (float*)(ws + 117964800);
  float* omi = (float*)(ws + 118226944);
  unsigned short* Bkan = (unsigned short*)(ws + 118489088);  // 40 MB bf16

  dim3 gg(16, 16);                             // 128x128 macro-tiles

  sym_kernel<<<2048, 256, 0, stream>>>(ids, tmap, gens, x);
  kan_prep_kernel<<<16384, 256, 0, stream>>>(kanc, Bkan);
  dft_kernel<<<256, 256, 0, stream>>>(x, xfr, xfi);
  spec_kernel<<<32, 64, 0, stream>>>(xfr, xfi, swr, swi, omr, omi);
  gemm_bf16_kernel<1, 0, 0><<<gg, 256, 0, stream>>>(x, (const void*)Bkan, x2,
                                                    2048, 2048, 10240, 2048);
  ifft_gelu_kernel<<<16384, 256, 0, stream>>>(omr, omi, x2, x);

  for (int l = 0; l < 4; ++l) {
    const size_t wo = (size_t)l * 2048 * 2048;
    rmsnorm_kernel<<<2048, 256, 0, stream>>>(x, normw + (size_t)l * 2048, hb);
    gemm_bf16_kernel<0, 1, 0><<<gg, 256, 0, stream>>>(hb, (const void*)(qw + wo), qb,
                                                      2048, 2048, 2048, 2048);
    gemm_bf16_kernel<0, 1, 0><<<gg, 256, 0, stream>>>(hb, (const void*)(kw + wo), kb,
                                                      2048, 2048, 2048, 2048);
    gemm_bf16_kernel<0, 1, 0><<<gg, 256, 0, stream>>>(hb, (const void*)(vw + wo), vb,
                                                      2048, 2048, 2048, 2048);
    rope_kernel<<<8192, 256, 0, stream>>>(qb);
    rope_kernel<<<8192, 256, 0, stream>>>(kb);
    attn_kernel<<<256, 256, 0, stream>>>(qb, kb, vb, comp + l * 16, phs + l * 16, ao);
    gemm_bf16_kernel<0, 1, 1><<<gg, 256, 0, stream>>>(ao, (const void*)(ow + wo), x,
                                                      2048, 2048, 2048, 2048);  // residual
  }
  rmsnorm_kernel<<<2048, 256, 0, stream>>>(x, fnw, out);
}
